// CounterRegression_13022340842124
// MI455X (gfx1250) — matrix operand order repaired and checked by reading, not compiled, not run on hardware
//
#include <hip/hip_runtime.h>

// CounterRegression: out = leaky_relu(cumsum(delta[input_seq], axis=0))
// input_seq: [T,B] int32 in {0,1};  delta: [2] f32;  out: [T,B] f32.
//
// counter[t,b] = delta0*(t+1) + (delta1-delta0)*prefix_ones[t,b]
// => column-wise integer prefix count, done hierarchically:
//   pass1: per-(chunk,col) bit sums
//   pass2: exclusive scan over chunks per column
//   pass3: intra-chunk prefix via triangular V_WMMA_F32_16X16X32_F16 + emit

#define T_DIM 8192
#define B_DIM 4096
#define CT 256                      // time steps per chunk
#define NCHUNK (T_DIM / CT)         // 32
#define NEG_SLOPE 0.01f

typedef __attribute__((ext_vector_type(16))) _Float16 v16h;
typedef __attribute__((ext_vector_type(8)))  float    v8f;

__global__ void __launch_bounds__(256)
cr_chunk_sum(const int* __restrict__ in, int* __restrict__ partial) {
    const int col   = blockIdx.x * blockDim.x + threadIdx.x;  // 0..B-1
    const int chunk = blockIdx.y;
    const int* p = in + (size_t)chunk * CT * B_DIM + col;
    int s = 0;
#pragma unroll 8
    for (int t = 0; t < CT; ++t) s += p[(size_t)t * B_DIM];
    partial[chunk * B_DIM + col] = s;
}

__global__ void __launch_bounds__(256)
cr_chunk_scan(const int* __restrict__ partial, int* __restrict__ offs) {
    const int col = blockIdx.x * blockDim.x + threadIdx.x;
    int acc = 0;
#pragma unroll
    for (int c = 0; c < NCHUNK; ++c) {
        offs[c * B_DIM + col] = acc;
        acc += partial[c * B_DIM + col];
    }
}

// One wave handles 16 columns x CT time steps. Per 32-step tile:
//   D1 = L1 * X + C   (prefixes for steps t0..t0+15)
//   D2 = L2 * X + C   (prefixes for steps t0+16..t0+31)
// C carries the running per-column offset (broadcast over all 8 acc VGPRs).
__global__ void __launch_bounds__(256)
cr_scan_wmma(const int* __restrict__ in, const int* __restrict__ offs,
             const float* __restrict__ delta, float* __restrict__ out) {
    const int lane  = threadIdx.x & 31;
    const int wave  = threadIdx.x >> 5;
    const int n     = lane & 15;          // column within 16-group (and M for A)
    const int hi    = lane >> 4;          // half-wave selector
    const int col   = blockIdx.x * 128 + wave * 16 + n;
    const int chunk = blockIdx.y;
    const int tBase = chunk * CT;

    const float d0 = delta[0];
    const float dd = delta[1] - d0;

    // Triangular operators, K order matched to B:
    // lane<16: elems0-15 <-> K0-15
    // lane>=16: elems0-15 <-> K16-31;  M = lane&15
    v16h a1, a2;
#pragma unroll
    for (int j = 0; j < 16; ++j) {
        const int k = hi * 16 + j;
        a1[j] = (_Float16)((k <= n)      ? 1.0f : 0.0f);   // L1[m,k] = k<=m
        a2[j] = (_Float16)((k <= n + 16) ? 1.0f : 0.0f);   // L2[m,k] = k<=m+16
    }

    float off = (float)offs[chunk * B_DIM + col];  // exclusive chunk offset (exact int)

    for (int t0 = 0; t0 < CT; t0 += 32) {
        // B layout: lanes 0-15 hold K=0..15, lanes 16-31 hold K=16..31, N = lane&15
        v16h b;
#pragma unroll
        for (int j = 0; j < 16; ++j) {
            const int t = tBase + t0 + hi * 16 + j;
            b[j] = (_Float16)(float)in[(size_t)t * B_DIM + col];  // bits 0/1, exact
        }

        v8f c;
#pragma unroll
        for (int r = 0; r < 8; ++r) c[r] = off;

        v8f dA = __builtin_amdgcn_wmma_f32_16x16x32_f16(false, a1, false, b,
                                                        (short)0, c, false, false);
        v8f dB = __builtin_amdgcn_wmma_f32_16x16x32_f16(false, a2, false, b,
                                                        (short)0, c, false, false);

        // Row 15 of D2 = offset + ones over all 32 steps; it lives in VGPR7,
        // lanes 16-31 (N = lane-16). Broadcast to both half-waves.
        off = __shfl(dB[7], 16 + n, 32);

        // C/D layout: VGPR r -> M = r + 8*hi, N = lane&15
#pragma unroll
        for (int r = 0; r < 8; ++r) {
            {
                const int tt = tBase + t0 + hi * 8 + r;
                float y = d0 * (float)(tt + 1) + dd * dA[r];
                y = (y >= 0.0f) ? y : NEG_SLOPE * y;
                __builtin_nontemporal_store(y, &out[(size_t)tt * B_DIM + col]);
            }
            {
                const int tt = tBase + t0 + 16 + hi * 8 + r;
                float y = d0 * (float)(tt + 1) + dd * dB[r];
                y = (y >= 0.0f) ? y : NEG_SLOPE * y;
                __builtin_nontemporal_store(y, &out[(size_t)tt * B_DIM + col]);
            }
        }
    }
}

extern "C" void kernel_launch(void* const* d_in, const int* in_sizes, int n_in,
                              void* d_out, int out_size, void* d_ws, size_t ws_size,
                              hipStream_t stream) {
    (void)in_sizes; (void)n_in; (void)out_size; (void)ws_size;
    const int*   in    = (const int*)d_in[0];
    const float* delta = (const float*)d_in[1];
    float*       out   = (float*)d_out;

    int* partial = (int*)d_ws;                 // NCHUNK*B ints = 512 KiB
    int* offs    = partial + NCHUNK * B_DIM;   // NCHUNK*B ints = 512 KiB

    cr_chunk_sum <<<dim3(B_DIM / 256, NCHUNK), 256, 0, stream>>>(in, partial);
    cr_chunk_scan<<<dim3(B_DIM / 256),         256, 0, stream>>>(partial, offs);
    cr_scan_wmma <<<dim3(B_DIM / 128, NCHUNK), 256, 0, stream>>>(in, offs, delta, out);
}
